// ResNetWSL_81767587381255
// MI455X (gfx1250) — matrix operand order repaired and checked by reading, not compiled, not run on hardware
//
#include <hip/hip_runtime.h>
#include <hip/hip_bf16.h>

// MI455X (gfx1250) fused ResNet-WSL head.
// Memory-bound: x read twice (411 MB) ~= 18us @ 23.3 TB/s; GEMMs pushed onto
// WMMA pipes (bf16 16x16x32 for the 1x1 conv, f32 16x16x4 for the classifier)
// so matrix math is negligible vs HBM traffic.

typedef __attribute__((ext_vector_type(8)))  float  v8f;
typedef __attribute__((ext_vector_type(2)))  float  v2f;
typedef __attribute__((ext_vector_type(16))) __bf16 v16bf;

#define NB   128
#define NC   2048
#define NP   196        // 14*14
#define NO   32         // nc*nm
#define NCLS 8
#define NMAP 4
#define MT_PER_B 13     // ceil(196/16)

// ---------------------------------------------------------------- k0: w_down f32 -> bf16
__global__ __launch_bounds__(256) void k_w2bf(const float* __restrict__ w,
                                              __bf16* __restrict__ wbf) {
    int i = blockIdx.x * 256 + threadIdx.x;      // 65536 total
    if (i < NO * NC) wbf[i] = (__bf16)w[i];
}

// ---------------------------------------------------------------- k1: f2 GEMM -> m_c
// One wave per (b, 16-pixel M-tile). A = x pixels (bf16-converted on the fly),
// B = w_down (bf16, preconverted), accumulate f32, emit m_c = mean over nm maps.
// Padded pixel rows (p >= 196) load clamped-but-valid addresses; their garbage
// D rows are simply never stored, so no per-element masking is needed.
__global__ __launch_bounds__(256) void k_gemm_mc(const float* __restrict__ x,
                                                 const __bf16* __restrict__ wbf,
                                                 const float* __restrict__ b_down,
                                                 float* __restrict__ mc_out) {
    __shared__ float sAcc[8][16 * NO];           // per-wave 16px x 32ch tile (16 KB)
    const int lane  = threadIdx.x & 31;
    const int wslot = threadIdx.x >> 5;
    const int wid   = blockIdx.x * 8 + wslot;    // 0..1663
    const int b     = wid / MT_PER_B;
    const int mt    = wid - b * MT_PER_B;
    const int p0    = mt * 16;
    const int lh    = lane >> 4;                 // lane half (0/1)
    const int lm    = lane & 15;
    const int pix   = p0 + lm;                   // A-matrix row = pixel
    const int pc    = (pix < NP) ? pix : (NP - 1);  // clamp OOB rows in-bounds
    const float* xb = x + (size_t)b * NC * NP;

    v8f acc0 = {}; v8f acc1 = {};
    #pragma unroll 2
    for (int kb = 0; kb < NC; kb += 32) {
        // A: 16x32 bf16, lane(lm,lh) reg r holds K = (r<4 ? 8*lh+2r : 16+8*lh+2(r-4)) (+j)
        v16bf a;
        #pragma unroll
        for (int r = 0; r < 8; ++r) {
            int k = kb + ((r < 4) ? (8 * lh + 2 * r) : (16 + 8 * lh + 2 * (r - 4)));
            float f0 = xb[(size_t)k * NP + pc];
            float f1 = xb[(size_t)(k + 1) * NP + pc];
            a[2 * r]     = (__bf16)f0;
            a[2 * r + 1] = (__bf16)f1;
        }
        // B: 32x16 bf16, lane(lm,lh) element e holds K = (e<8 ? 8*lh+e : 16+8*lh+(e-8)), N = lm (+16 for tile 1)
        typedef __attribute__((ext_vector_type(8))) __bf16 v8bf_;
        const __bf16* bp0 = wbf + (size_t)lm        * NC + kb + 8 * lh;
        const __bf16* bp1 = wbf + (size_t)(16 + lm) * NC + kb + 8 * lh;
        v16bf bm0 = __builtin_shufflevector(*(const v8bf_*)bp0, *(const v8bf_*)(bp0 + 16), 0,1,2,3,4,5,6,7,8,9,10,11,12,13,14,15);
        v16bf bm1 = __builtin_shufflevector(*(const v8bf_*)bp1, *(const v8bf_*)(bp1 + 16), 0,1,2,3,4,5,6,7,8,9,10,11,12,13,14,15);
        acc0 = __builtin_amdgcn_wmma_f32_16x16x32_bf16(false, a, false, bm0,
                                                       (short)0, acc0, false, false);
        acc1 = __builtin_amdgcn_wmma_f32_16x16x32_bf16(false, a, false, bm1,
                                                       (short)0, acc1, false, false);
    }
    const float bia0 = b_down[lm];
    const float bia1 = b_down[16 + lm];
    // D layout: element (M = r + 8*lh, N = lm)
    #pragma unroll
    for (int r = 0; r < 8; ++r) {
        int pr = r + 8 * lh;
        sAcc[wslot][pr * NO + lm]      = acc0[r] + bia0;
        sAcc[wslot][pr * NO + 16 + lm] = acc1[r] + bia1;
    }
    __syncthreads();
    // ClassWisePool over nm=4 -> m_c; 16px * 8cls = 128 values per wave
    #pragma unroll
    for (int i = 0; i < 4; ++i) {
        int idx = lane + 32 * i;                 // 0..127
        int pr  = idx >> 3;
        int cls = idx & 7;
        int p   = p0 + pr;
        if (p < NP) {
            const float* row = &sAcc[wslot][pr * NO + 4 * cls];
            float mval = 0.25f * (row[0] + row[1] + row[2] + row[3]);
            mc_out[((size_t)b * NCLS + cls) * NP + p] = mval;
        }
    }
}

// ---------------------------------------------------------------- k2: v (= pixel-mean of m_c) and m
__global__ __launch_bounds__(256) void k_vm(const float* __restrict__ mc,
                                            float* __restrict__ v_out,
                                            float* __restrict__ m_out) {
    __shared__ float s_mc[NCLS * NP];
    __shared__ float s_v[NCLS];
    const int b = blockIdx.x;
    const float* mcb = mc + (size_t)b * NCLS * NP;
    for (int i = threadIdx.x; i < NCLS * NP; i += 256) s_mc[i] = mcb[i];
    __syncthreads();
    if (threadIdx.x < NCLS) {
        float s = 0.0f;
        for (int p = 0; p < NP; ++p) s += s_mc[threadIdx.x * NP + p];
        float vv = s * (1.0f / (float)NP);
        v_out[b * NCLS + threadIdx.x] = vv;
        s_v[threadIdx.x] = vv;
    }
    __syncthreads();
    for (int p = threadIdx.x; p < NP; p += 256) {
        float s = 0.0f;
        #pragma unroll
        for (int c = 0; c < NCLS; ++c) s += s_v[c] * s_mc[c * NP + p];
        m_out[(size_t)b * NP + p] = s * (1.0f / (float)NCLS);
    }
}

// ---------------------------------------------------------------- k3: streaming d = [mean x, mean m*x]
// One wave per (b, channel): 784 contiguous bytes per wave, deterministic shfl reduce.
__global__ __launch_bounds__(256) void k_du(const float* __restrict__ x,
                                            const float* __restrict__ m,
                                            float* __restrict__ dvec) {
    const int wid  = blockIdx.x * 8 + (threadIdx.x >> 5);  // 0..262143
    const int lane = threadIdx.x & 31;
    const int b = wid >> 11;
    const int c = wid & (NC - 1);
    const float* xr = x + ((size_t)b * NC + c) * NP;
    const float* mr = m + (size_t)b * NP;
    float xs = 0.0f, us = 0.0f;
    for (int p = lane; p < NP; p += 32) {
        __builtin_prefetch(xr + p + 64, 0, 1);   // global_prefetch_b8 (speculative OK)
        float xv = xr[p];
        float mv = mr[p];
        xs += xv;
        us += mv * xv;
    }
    #pragma unroll
    for (int off = 16; off > 0; off >>= 1) {
        xs += __shfl_xor(xs, off, 32);
        us += __shfl_xor(us, off, 32);
    }
    if (lane == 0) {
        dvec[(size_t)b * (2 * NC) + c]      = xs * (1.0f / (float)NP);
        dvec[(size_t)b * (2 * NC) + NC + c] = us * (1.0f / (float)NP);
    }
}

// ---------------------------------------------------------------- k4: classifier d[128,4096] @ w_cls^T
#if __has_builtin(__builtin_amdgcn_wmma_f32_16x16x4_f32)
__global__ __launch_bounds__(256) void k_cls(const float* __restrict__ dvec,
                                             const float* __restrict__ w_cls,
                                             const float* __restrict__ b_cls,
                                             float* __restrict__ out) {
    const int lane = threadIdx.x & 31;
    const int wv   = threadIdx.x >> 5;           // m-tile 0..7 (rows 16wv..16wv+15)
    const int lh   = lane >> 4;
    const int lm   = lane & 15;
    const bool nv  = lm < NCLS;                  // pad N 8..15 with zeros
    const float* arow = dvec + (size_t)(wv * 16 + lm) * (2 * NC);
    const float* brow = w_cls + (size_t)(nv ? lm : 0) * (2 * NC);
    v8f acc = {};
    for (int kb = 0; kb < 2 * NC; kb += 4) {
        // f32 A 16x4: reg r holds K = 2*lh + r; B 4x16 mirrors; both b64 loads
        v2f a  = *(const v2f*)(arow + kb + 2 * lh);
        v2f bb = *(const v2f*)(brow + kb + 2 * lh);
        bb[0] = nv ? bb[0] : 0.0f;
        bb[1] = nv ? bb[1] : 0.0f;
        acc = __builtin_amdgcn_wmma_f32_16x16x4_f32(false, a, false, bb,
                                                    (short)0, acc, false, false);
    }
    if (nv) {
        float bias = b_cls[lm];
        #pragma unroll
        for (int r = 0; r < 8; ++r) {
            int row = wv * 16 + r + 8 * lh;      // D: M = r + 8*lh
            out[row * NCLS + lm] = acc[r] + bias;
        }
    }
}
#else
__global__ __launch_bounds__(256) void k_cls(const float* __restrict__ dvec,
                                             const float* __restrict__ w_cls,
                                             const float* __restrict__ b_cls,
                                             float* __restrict__ out) {
    for (int idx = threadIdx.x; idx < NB * NCLS; idx += 256) {
        int row = idx >> 3, n = idx & 7;
        const float* a = dvec + (size_t)row * (2 * NC);
        const float* w = w_cls + (size_t)n * (2 * NC);
        float s = 0.0f;
        for (int k = 0; k < 2 * NC; ++k) s += a[k] * w[k];
        out[idx] = s + b_cls[n];
    }
}
#endif

// ---------------------------------------------------------------- launch
extern "C" void kernel_launch(void* const* d_in, const int* in_sizes, int n_in,
                              void* d_out, int out_size, void* d_ws, size_t ws_size,
                              hipStream_t stream) {
    const float* x      = (const float*)d_in[0];
    const float* w_down = (const float*)d_in[1];
    const float* b_down = (const float*)d_in[2];
    const float* w_cls  = (const float*)d_in[3];
    const float* b_cls  = (const float*)d_in[4];

    float* out     = (float*)d_out;
    float* v_out   = out;                        // [128,8]        offset 0
    float* cls_out = out + 1024;                 // [128,8]        offset 1024
    float* m_out   = out + 2048;                 // [128,1,14,14]  offset 2048
    float* mc_out  = out + 27136;                // [128,8,14,14]  offset 27136

    __bf16* wbf = (__bf16*)d_ws;                         // 128 KB
    float*  dvec = (float*)((char*)d_ws + 131072);       // 2 MB: d[128,4096]

    k_w2bf   <<<dim3(256),   dim3(256), 0, stream>>>(w_down, wbf);
    k_gemm_mc<<<dim3(208),   dim3(256), 0, stream>>>(x, wbf, b_down, mc_out);
    k_vm     <<<dim3(NB),    dim3(256), 0, stream>>>(mc_out, v_out, m_out);
    k_du     <<<dim3(32768), dim3(256), 0, stream>>>(x, m_out, dvec);
    k_cls    <<<dim3(1),     dim3(256), 0, stream>>>(dvec, w_cls, b_cls, cls_out);
}
